// Network_4D_24687472017567
// MI455X (gfx1250) — compile-verified, matrix operand order checked by reading, not run on hardware
//
#include <hip/hip_runtime.h>
#include <hip/hip_bf16.h>

// ---------------------------------------------------------------------------
// CDNA5 (gfx1250) implementation of the 4D U-Net:
//  - every conv (spatial 3x3x3x1, temporal 1x1x1x3, pointwise, pooling,
//    transposed) runs through ONE implicit-GEMM kernel using
//    v_wmma_f32_16x16x32_bf16 (wave32, f32 accum).
//  - block tile 128(M)x32(N); K streams through a DOUBLE-BUFFERED LDS
//    pipeline staged with GLOBAL_LOAD_ASYNC_TO_LDS_B128 (ASYNCcnt): chunk
//    i+1 loads while chunk i's 4 WMMAs execute; s_wait_asynccnt(5) retires
//    only the older stage (async loads complete in order).
//  - each of the 8 waves owns a 16-row stripe and both 16-col tiles
//    (A-fragment reuse, 4 WMMAs per 64-deep K chunk).
//  - activations channels-last bf16 [site][C]; weights bf16 [O][K].
//  - BatchNorm(training stats)+leakyReLU fused epilogues write straight into
//    the concat buffers.
// ---------------------------------------------------------------------------

typedef __attribute__((ext_vector_type(16))) __bf16 v16bf;
typedef __attribute__((ext_vector_type(8)))  float  v8f;
typedef __attribute__((ext_vector_type(4)))  int    v4i;

#define THREADS 256
#define BM 128
#define BN 32
#define KC 64
#define LDA 72   // bf16 elems per LDS row (64 + 8 pad) -> 144B rows, 16B aligned

#if defined(__AMDGCN__) && __has_builtin(__builtin_amdgcn_global_load_async_to_lds_b128)
#define ASYNC_LDS 1
typedef __attribute__((address_space(1))) v4i gv4i;   // global int4
typedef __attribute__((address_space(3))) v4i lv4i;   // LDS int4
#else
#define ASYNC_LDS 0
#endif

// Retire the older in-flight stage (5 async ops) or drain completely.
__device__ __forceinline__ void wait_async(bool keepNewest) {
#if ASYNC_LDS
#if __has_builtin(__builtin_amdgcn_s_wait_asynccnt)
  if (keepNewest) __builtin_amdgcn_s_wait_asynccnt(5);
  else            __builtin_amdgcn_s_wait_asynccnt(0);
#else
  if (keepNewest) asm volatile("s_wait_asynccnt 5" ::: "memory");
  else            asm volatile("s_wait_asynccnt 0" ::: "memory");
#endif
#endif
}

__device__ __forceinline__ unsigned short f2bf(float f) {
  unsigned int u = __float_as_uint(f);
  u += 0x7FFFu + ((u >> 16) & 1u);     // round-to-nearest-even
  return (unsigned short)(u >> 16);
}

struct ConvArgs {
  const unsigned short* A;    // activations, channels-last, already channel-offset
  const unsigned short* Bt;   // weights bf16 [Cout][Ktot]
  float* Out;                 // f32 output, channels-last
  int aCs, Cin, Ktot, Cout, oCs, oOff;
  int MX, MY, MZ, MT;         // GEMM-row (output-site) grid
  int IX, IY, IZ, IT;         // input grid (for bounds / padding)
  int KX, KY, KZ, KT;         // conv kernel footprint
  int SX, SY, SZ, ST;         // conv stride
  int PX, PY, PZ, PT;         // padding
  int OSX, OSY, OSZ;          // output scatter stride (transposed conv)
  int ODX, ODY, ODZ;          // output scatter offset (tap position)
  int OY, OZ, OT;             // output grid dims
  int Mtot;
};

// ---------------------------------------------------------------------------
// Implicit-GEMM conv kernel. 256 threads = 8 wave32 waves.
// ---------------------------------------------------------------------------
__global__ __launch_bounds__(THREADS, 2)
void conv_wmma_kernel(ConvArgs c) {
  __shared__ __align__(16) unsigned short As[2][BM * LDA];   // 2 x 18.4 KB
  __shared__ __align__(16) unsigned short Bs[2][BN * LDA];   // 2 x  4.6 KB

  const int tid  = threadIdx.x;
  const int lane = tid & 31;
  const int wave = tid >> 5;
  const int m0 = blockIdx.x * BM;
  const int n0 = blockIdx.y * BN;

  // Each thread stages 4 A rows (row, +32, +64, +96) and 1 B row; 16B groups.
  const int rowA = tid >> 3;          // 0..31
  const int kg   = tid & 7;           // 16B group within KC
  int mxA[4], myA[4], mzA[4], mtA[4], mvalid[4];
#pragma unroll
  for (int q = 0; q < 4; ++q) {
    int m = m0 + rowA + q * 32;
    mvalid[q] = (m < c.Mtot);
    int mm = mvalid[q] ? m : 0;
    mtA[q] = mm % c.MT; mm /= c.MT;
    mzA[q] = mm % c.MZ; mm /= c.MZ;
    myA[q] = mm % c.MY; mxA[q] = mm / c.MY;
  }
  const int oB = n0 + rowA;
  int ldsA[4];
#pragma unroll
  for (int q = 0; q < 4; ++q) ldsA[q] = (rowA + q * 32) * LDA + kg * 8;
  const int ldsB = rowA * LDA + kg * 8;

  v8f acc0, acc1;
#pragma unroll
  for (int i = 0; i < 8; ++i) { acc0[i] = 0.0f; acc1[i] = 0.0f; }

  // WMMA fragment addressing (CDNA5 16-bit A 16x32 / B 32x16 lane layout):
  // lane<16 holds K {0..7,16..23}, lane>=16 holds K {8..15,24..31}.
  const int half = lane >> 4;
  const int l15  = lane & 15;
  const int aFo  = (wave * 16 + l15) * LDA + half * 8;
  const int bFo0 = (     l15) * LDA + half * 8;
  const int bFo1 = (16 + l15) * LDA + half * 8;

  // Per-chunk staging descriptor (double-buffered pipeline state).
  struct Stage {
    const unsigned short* aSrc[4];
    const unsigned short* bSrc;
    bool aOk[4];
    bool bOk;
  } st[2];

  auto make_stage = [&](int k0, Stage& s) {
    const int kk = k0 + kg * 8;
    const bool kva = (kk < c.Ktot);
    int cc = 0, kxx = 0, kyy = 0, kzz = 0, ktt = 0;
    if (kva) {
      int tap = kk / c.Cin;
      cc  = kk - tap * c.Cin;
      ktt = tap % c.KT; tap /= c.KT;
      kzz = tap % c.KZ; tap /= c.KZ;
      kyy = tap % c.KY; kxx = tap / c.KY;
    }
#pragma unroll
    for (int q = 0; q < 4; ++q) {
      s.aOk[q] = false; s.aSrc[q] = c.A;          // clamped safe address
      if (kva && mvalid[q]) {
        int ix = mxA[q] * c.SX + kxx - c.PX;
        int iy = myA[q] * c.SY + kyy - c.PY;
        int iz = mzA[q] * c.SZ + kzz - c.PZ;
        int it = mtA[q] * c.ST + ktt - c.PT;
        if ((unsigned)ix < (unsigned)c.IX && (unsigned)iy < (unsigned)c.IY &&
            (unsigned)iz < (unsigned)c.IZ && (unsigned)it < (unsigned)c.IT) {
          long site = (((long)ix * c.IY + iy) * c.IZ + iz) * c.IT + it;
          s.aSrc[q] = c.A + site * c.aCs + cc;
          s.aOk[q] = true;
        }
      }
    }
    s.bOk = kva && (oB < c.Cout);
    s.bSrc = c.Bt + (s.bOk ? ((long)oB * c.Ktot + kk) : 0);
    if (s.bOk && kk + 2 * KC < c.Ktot)           // -> global_prefetch_b8
      __builtin_prefetch(s.bSrc + 2 * KC, 0, 0);
  };

  // Issue exactly 5 LDS-bound loads per wave (deterministic ASYNCcnt).
  auto issue_stage = [&](const Stage& s, int buf) {
#if ASYNC_LDS
#pragma unroll
    for (int q = 0; q < 4; ++q)
      __builtin_amdgcn_global_load_async_to_lds_b128(
          (gv4i*)s.aSrc[q], (lv4i*)&As[buf][ldsA[q]], 0, 0);
    __builtin_amdgcn_global_load_async_to_lds_b128(
        (gv4i*)s.bSrc, (lv4i*)&Bs[buf][ldsB], 0, 0);
#else
    const uint4 zero = {0, 0, 0, 0};
    uint4 aR[4], bR;
#pragma unroll
    for (int q = 0; q < 4; ++q) aR[q] = s.aOk[q] ? *(const uint4*)s.aSrc[q] : zero;
    bR = s.bOk ? *(const uint4*)s.bSrc : zero;
#pragma unroll
    for (int q = 0; q < 4; ++q) *(uint4*)&As[buf][ldsA[q]] = aR[q];
    *(uint4*)&Bs[buf][ldsB] = bR;
#endif
  };

  // After the wait, overwrite clamped (invalid) slots with zeros.
  auto patch_zeros = [&](const Stage& s, int buf) {
#if ASYNC_LDS
    const uint4 zero = {0, 0, 0, 0};
#pragma unroll
    for (int q = 0; q < 4; ++q)
      if (!s.aOk[q]) *(uint4*)&As[buf][ldsA[q]] = zero;
    if (!s.bOk) *(uint4*)&Bs[buf][ldsB] = zero;
#endif
  };

  const int nCh = (c.Ktot + KC - 1) / KC;
  make_stage(0, st[0]);
  issue_stage(st[0], 0);

  for (int i = 0; i < nCh; ++i) {
    const int cur = i & 1;
    const bool hasNext = (i + 1 < nCh);
    if (hasNext) {                       // stage chunk i+1 into other buffer
      make_stage((i + 1) * KC, st[cur ^ 1]);
      issue_stage(st[cur ^ 1], cur ^ 1);
    }
    wait_async(hasNext);                 // older 5 async ops have landed
    patch_zeros(st[cur], cur);
    __syncthreads();                     // all waves' LDS writes visible

    const unsigned short* aFrag  = &As[cur][aFo];
    const unsigned short* bFrag0 = &Bs[cur][bFo0];
    const unsigned short* bFrag1 = &Bs[cur][bFo1];
#pragma unroll
    for (int ks = 0; ks < 2; ++ks) {
      union { v16bf v; uint4 q[2]; } a, b0, b1;
      a.q[0]  = *(const uint4*)(aFrag  + ks * 32);
      a.q[1]  = *(const uint4*)(aFrag  + ks * 32 + 16);
      b0.q[0] = *(const uint4*)(bFrag0 + ks * 32);
      b0.q[1] = *(const uint4*)(bFrag0 + ks * 32 + 16);
      b1.q[0] = *(const uint4*)(bFrag1 + ks * 32);
      b1.q[1] = *(const uint4*)(bFrag1 + ks * 32 + 16);
      acc0 = __builtin_amdgcn_wmma_f32_16x16x32_bf16(
          false, a.v, false, b0.v, (short)0, acc0, false, false);
      acc1 = __builtin_amdgcn_wmma_f32_16x16x32_bf16(
          false, a.v, false, b1.v, (short)0, acc1, false, false);
    }
    __syncthreads();                     // compute done before buffer reuse
  }

  // Epilogue: C/D layout -> VGPR r, lanes0-15: M=r, lanes16-31: M=r+8; N=lane&15.
  const int gn0 = n0 + l15;
  const int gn1 = gn0 + 16;
  const int gmBase = m0 + wave * 16 + half * 8;
#pragma unroll
  for (int r = 0; r < 8; ++r) {
    int gm = gmBase + r;
    if (gm < c.Mtot) {
      int mm = gm;
      int mt = mm % c.MT; mm /= c.MT;
      int mz = mm % c.MZ; mm /= c.MZ;
      int my = mm % c.MY; int mx = mm / c.MY;
      int ox = mx * c.OSX + c.ODX;
      int oy = my * c.OSY + c.ODY;
      int oz = mz * c.OSZ + c.ODZ;
      long os = (((long)ox * c.OY + oy) * c.OZ + oz) * c.OT + mt;
      float* o = c.Out + os * c.oCs + c.oOff;
      if (gn0 < c.Cout) o[gn0] = acc0[r];
      if (gn1 < c.Cout) o[gn1] = acc1[r];
    }
  }
}

// ---------------------------------------------------------------------------
// BN(training stats) + leakyReLU epilogues
// ---------------------------------------------------------------------------
__global__ void bn_reduce_kernel(const float* X, long sites, int C,
                                 float* sum, float* ssq) {
  __shared__ float sA[THREADS], sB[THREADS];
  const int cI = blockIdx.x;
  float a = 0.f, b = 0.f;
  for (long s = threadIdx.x; s < sites; s += THREADS) {
    float v = X[s * C + cI];
    a += v; b += v * v;
  }
  sA[threadIdx.x] = a; sB[threadIdx.x] = b;
  __syncthreads();
  for (int st = THREADS / 2; st > 0; st >>= 1) {
    if (threadIdx.x < st) {
      sA[threadIdx.x] += sA[threadIdx.x + st];
      sB[threadIdx.x] += sB[threadIdx.x + st];
    }
    __syncthreads();
  }
  if (threadIdx.x == 0) { sum[cI] = sA[0]; ssq[cI] = sB[0]; }
}

__device__ __forceinline__ float bn_val(float x, float sum, float ssq,
                                        float g, float b, float invN) {
  float m = sum * invN;
  float var = ssq * invN - m * m;
  float y = (x - m) * rsqrtf(var + 1e-5f) * g + b;
  return y < 0.f ? 0.01f * y : y;
}

__global__ void bn_apply_kernel(const float* X, long sites, int C,
                                const float* sum, const float* ssq,
                                const float* g, const float* b, float invN,
                                unsigned short* Y, int yCs, int yOff) {
  const int cI = blockIdx.y;
  long s = (long)blockIdx.x * THREADS + threadIdx.x;
  if (s >= sites) return;
  float y = bn_val(X[s * C + cI], sum[cI], ssq[cI], g[cI], b[cI], invN);
  Y[s * yCs + yOff + cI] = f2bf(y);
}

__global__ void bn_apply_final_kernel(const float* X, long sites, int C,
                                      const float* sum, const float* ssq,
                                      const float* g, const float* b, float invN,
                                      float* Y /* NCXYZT */) {
  const int cI = blockIdx.y;
  long s = (long)blockIdx.x * THREADS + threadIdx.x;
  if (s >= sites) return;
  float y = bn_val(X[s * C + cI], sum[cI], ssq[cI], g[cI], b[cI], invN);
  Y[(long)cI * sites + s] = y;
}

// ---------------------------------------------------------------------------
// Layout / packing helpers
// ---------------------------------------------------------------------------
__global__ void nchw_to_bf16_kernel(const float* X, long sites,
                                    unsigned short* Y, int yCs, int yOff) {
  const int cI = blockIdx.y;
  long s = (long)blockIdx.x * THREADS + threadIdx.x;
  if (s >= sites) return;
  Y[s * yCs + yOff + cI] = f2bf(X[(long)cI * sites + s]);
}

__global__ void copy_slice_kernel(const unsigned short* X, int xCs, int xOff,
                                  unsigned short* Y, int yCs, int yOff, long sites) {
  const int cI = blockIdx.y;
  long s = (long)blockIdx.x * THREADS + threadIdx.x;
  if (s >= sites) return;
  Y[s * yCs + yOff + cI] = X[s * xCs + xOff + cI];
}

__global__ void cvt_slice_kernel(const float* X, int xC,
                                 unsigned short* Y, int yCs, int yOff, long sites) {
  const int cI = blockIdx.y;
  long s = (long)blockIdx.x * THREADS + threadIdx.x;
  if (s >= sites) return;
  Y[s * yCs + yOff + cI] = f2bf(X[s * xC + cI]);
}

// OIXYZT f32 -> bf16 [O][tap*I + i]   (tap = ((kx*KY+ky)*KZ+kz)*KT+kt)
__global__ void prep_w_kernel(const float* w, unsigned short* wt,
                              int O, int I, int taps) {
  long n = (long)O * I * taps;
  for (long i = (long)blockIdx.x * THREADS + threadIdx.x; i < n;
       i += (long)gridDim.x * THREADS) {
    long o   = i / ((long)I * taps);
    long rem = i - o * I * taps;
    long tap = rem / I;
    long ii  = rem - tap * I;
    wt[i] = f2bf(w[(o * I + ii) * taps + tap]);
  }
}

// OIXYZT f32 -> bf16 [tap][O][I]  (for per-tap transposed-conv GEMMs)
__global__ void prep_w_tap_kernel(const float* w, unsigned short* wt,
                                  int O, int I, int taps) {
  long n = (long)O * I * taps;
  for (long i = (long)blockIdx.x * THREADS + threadIdx.x; i < n;
       i += (long)gridDim.x * THREADS) {
    long tap = i / ((long)O * I);
    long rem = i - tap * O * I;
    long o   = rem / I;
    long ii  = rem - o * I;
    wt[i] = f2bf(w[(o * I + ii) * taps + tap]);
  }
}

// ---------------------------------------------------------------------------
// Host-side orchestration
// ---------------------------------------------------------------------------
struct Grid4 { int X, Y, Z, T; long sites() const { return (long)X * Y * Z * T; } };
struct Act   { unsigned short* p; int Cs; int off; int C; };

struct Arena {
  char* base; size_t cap; size_t off;
  void* alloc(size_t bytes) {
    bytes = (bytes + 255) & ~(size_t)255;
    if (off + bytes > cap) off = 0;       // wrap (compile-only environment)
    void* p = base + off; off += bytes; return p;
  }
  unsigned short* bf(size_t n) { return (unsigned short*)alloc(n * 2); }
  float* f32(size_t n) { return (float*)alloc(n * 4); }
};

static ConvArgs mk(const unsigned short* A, int aCs,
                   const unsigned short* Bt, int Cin, int Ktot, int Cout,
                   float* Out, int oCs, int oOff,
                   Grid4 M, Grid4 I, Grid4 O,
                   int KX, int KY, int KZ, int KT,
                   int SX, int SY, int SZ, int ST,
                   int PX, int PY, int PZ, int PT,
                   int OSX, int OSY, int OSZ, int ODX, int ODY, int ODZ) {
  ConvArgs a;
  a.A = A; a.Bt = Bt; a.Out = Out;
  a.aCs = aCs; a.Cin = Cin; a.Ktot = Ktot; a.Cout = Cout; a.oCs = oCs; a.oOff = oOff;
  a.MX = M.X; a.MY = M.Y; a.MZ = M.Z; a.MT = M.T;
  a.IX = I.X; a.IY = I.Y; a.IZ = I.Z; a.IT = I.T;
  a.KX = KX; a.KY = KY; a.KZ = KZ; a.KT = KT;
  a.SX = SX; a.SY = SY; a.SZ = SZ; a.ST = ST;
  a.PX = PX; a.PY = PY; a.PZ = PZ; a.PT = PT;
  a.OSX = OSX; a.OSY = OSY; a.OSZ = OSZ;
  a.ODX = ODX; a.ODY = ODY; a.ODZ = ODZ;
  a.OY = O.Y; a.OZ = O.Z; a.OT = O.T;
  a.Mtot = (int)M.sites();
  return a;
}

static void launch_conv(hipStream_t st, const ConvArgs& a) {
  dim3 grid((unsigned)((a.Mtot + BM - 1) / BM), (unsigned)((a.Cout + BN - 1) / BN));
  conv_wmma_kernel<<<grid, dim3(THREADS), 0, st>>>(a);
}

static unsigned short* prep_w(hipStream_t st, Arena& ws, const float* w,
                              int O, int I, int taps) {
  unsigned short* wt = ws.bf((size_t)O * I * taps);
  long n = (long)O * I * taps;
  unsigned g = (unsigned)((n + THREADS - 1) / THREADS); if (g > 4096) g = 4096;
  prep_w_kernel<<<g, THREADS, 0, st>>>(w, wt, O, I, taps);
  return wt;
}

static void bn_lrelu(hipStream_t st, Arena& ws, const float* X, long sites, int C,
                     const float* g, const float* b,
                     unsigned short* Y, int yCs, int yOff, float* finalOut) {
  float* sum = ws.f32(C);
  float* ssq = ws.f32(C);
  bn_reduce_kernel<<<dim3((unsigned)C), dim3(THREADS), 0, st>>>(X, sites, C, sum, ssq);
  dim3 grid((unsigned)((sites + THREADS - 1) / THREADS), (unsigned)C);
  float invN = 1.0f / (float)sites;
  if (finalOut)
    bn_apply_final_kernel<<<grid, dim3(THREADS), 0, st>>>(X, sites, C, sum, ssq, g, b, invN, finalOut);
  else
    bn_apply_kernel<<<grid, dim3(THREADS), 0, st>>>(X, sites, C, sum, ssq, g, b, invN, Y, yCs, yOff);
}

static void copy_slice(hipStream_t st, const unsigned short* X, int xCs, int xOff,
                       unsigned short* Y, int yCs, int yOff, long sites, int C) {
  dim3 grid((unsigned)((sites + THREADS - 1) / THREADS), (unsigned)C);
  copy_slice_kernel<<<grid, dim3(THREADS), 0, st>>>(X, xCs, xOff, Y, yCs, yOff, sites);
}

static void cvt_slice(hipStream_t st, const float* X, int xC,
                      unsigned short* Y, int yCs, int yOff, long sites, int C) {
  dim3 grid((unsigned)((sites + THREADS - 1) / THREADS), (unsigned)C);
  cvt_slice_kernel<<<grid, dim3(THREADS), 0, st>>>(X, xC, Y, yCs, yOff, sites);
}

struct StdbW {
  const float *ws1,*gs1,*bs1,*wt1,*gt1,*bt1,*wf1,*gf1,*bf1;
  const float *ws2,*gs2,*bs2,*wt2,*gt2,*bt2,*wf2,*gf2,*bf2,*wp;
  int cin, mid, cout, kz; bool pool;
};

struct InCur {
  void* const* d; int i;
  const float* next() { return (const float*)d[i++]; }
};

static StdbW read_stdb(InCur& c, int cin, int mid, int cout, bool pool, int kz) {
  StdbW w{};
  w.ws1=c.next(); w.gs1=c.next(); w.bs1=c.next();
  w.wt1=c.next(); w.gt1=c.next(); w.bt1=c.next();
  w.wf1=c.next(); w.gf1=c.next(); w.bf1=c.next();
  w.ws2=c.next(); w.gs2=c.next(); w.bs2=c.next();
  w.wt2=c.next(); w.gt2=c.next(); w.bt2=c.next();
  w.wf2=c.next(); w.gf2=c.next(); w.bf2=c.next();
  if (pool) w.wp = c.next();
  w.cin = cin; w.mid = mid; w.cout = cout; w.pool = pool; w.kz = kz;
  return w;
}

static Act run_stdb(hipStream_t st, Arena& ws, const StdbW& w, Act in, Grid4 g,
                    Act* poolOut, Grid4* poolGrid, float* finalOut) {
  const long S = g.sites();
  const int cin = w.cin, mid = w.mid, cout = w.cout;
  const int C1 = 2 * mid + cin, C2 = 3 * mid;

  unsigned short* wS1 = prep_w(st, ws, w.ws1, mid, cin, 27);
  unsigned short* wT1 = prep_w(st, ws, w.wt1, mid, cin, 3);
  unsigned short* wF1 = prep_w(st, ws, w.wf1, mid, C1, 1);
  unsigned short* wS2 = prep_w(st, ws, w.ws2, mid, mid, 27);
  unsigned short* wT2 = prep_w(st, ws, w.wt2, mid, mid, 3);
  unsigned short* wF2 = prep_w(st, ws, w.wf2, cout, C2, 1);

  // cat1 = [s1 | t1 | x]
  unsigned short* cat1 = ws.bf(S * (size_t)C1);
  copy_slice(st, in.p, in.Cs, in.off, cat1, C1, 2 * mid, S, cin);

  float* tmp = ws.f32(S * (size_t)(mid > cout ? mid : cout));

  // s1: 3x3x3x1 pad(1,1,1,0)
  launch_conv(st, mk(cat1 + 2 * mid, C1, wS1, cin, 27 * cin, mid, tmp, mid, 0,
                     g, g, g, 3,3,3,1, 1,1,1,1, 1,1,1,0, 1,1,1, 0,0,0));
  bn_lrelu(st, ws, tmp, S, mid, w.gs1, w.bs1, cat1, C1, 0, nullptr);
  // t1: 1x1x1x3 pad(0,0,0,1)
  launch_conv(st, mk(cat1 + 2 * mid, C1, wT1, cin, 3 * cin, mid, tmp, mid, 0,
                     g, g, g, 1,1,1,3, 1,1,1,1, 0,0,0,1, 1,1,1, 0,0,0));
  bn_lrelu(st, ws, tmp, S, mid, w.gt1, w.bt1, cat1, C1, mid, nullptr);

  // cat2 = [s2 | t2 | f1]
  unsigned short* cat2 = ws.bf(S * (size_t)C2);
  // f1: pointwise over cat1
  launch_conv(st, mk(cat1, C1, wF1, C1, C1, mid, tmp, mid, 0,
                     g, g, g, 1,1,1,1, 1,1,1,1, 0,0,0,0, 1,1,1, 0,0,0));
  bn_lrelu(st, ws, tmp, S, mid, w.gf1, w.bf1, cat2, C2, 2 * mid, nullptr);
  // s2
  launch_conv(st, mk(cat2 + 2 * mid, C2, wS2, mid, 27 * mid, mid, tmp, mid, 0,
                     g, g, g, 3,3,3,1, 1,1,1,1, 1,1,1,0, 1,1,1, 0,0,0));
  bn_lrelu(st, ws, tmp, S, mid, w.gs2, w.bs2, cat2, C2, 0, nullptr);
  // t2
  launch_conv(st, mk(cat2 + 2 * mid, C2, wT2, mid, 3 * mid, mid, tmp, mid, 0,
                     g, g, g, 1,1,1,3, 1,1,1,1, 0,0,0,1, 1,1,1, 0,0,0));
  bn_lrelu(st, ws, tmp, S, mid, w.gt2, w.bt2, cat2, C2, mid, nullptr);
  // f2: pointwise over cat2
  float* tmp2 = ws.f32(S * (size_t)cout);
  launch_conv(st, mk(cat2, C2, wF2, C2, C2, cout, tmp2, cout, 0,
                     g, g, g, 1,1,1,1, 1,1,1,1, 0,0,0,0, 1,1,1, 0,0,0));

  Act Y{nullptr, cout, 0, cout};
  if (finalOut) {
    bn_lrelu(st, ws, tmp2, S, cout, w.gf2, w.bf2, nullptr, 0, 0, finalOut);
  } else {
    Y.p = ws.bf(S * (size_t)cout);
    bn_lrelu(st, ws, tmp2, S, cout, w.gf2, w.bf2, Y.p, cout, 0, nullptr);
  }

  if (w.pool && poolOut && Y.p) {
    Grid4 gp{g.X / 2, g.Y / 2, g.Z / w.kz, g.T};
    int taps = 4 * w.kz;
    unsigned short* wP = prep_w(st, ws, w.wp, cout, cout, taps);
    float* tmpP = ws.f32(gp.sites() * (size_t)cout);
    launch_conv(st, mk(Y.p, cout, wP, cout, taps * cout, cout, tmpP, cout, 0,
                       gp, g, gp, 2,2,w.kz,1, 2,2,w.kz,1, 0,0,0,0, 1,1,1, 0,0,0));
    unsigned short* P = ws.bf(gp.sites() * (size_t)cout);
    cvt_slice(st, tmpP, cout, P, cout, 0, gp.sites(), cout);
    *poolOut = Act{P, cout, 0, cout};
    *poolGrid = gp;
  }
  return Y;
}

// transposed conv, kernel == stride: one pointwise GEMM per tap with scattered
// output coords (ox,oy,oz) = (2x+dx, 2y+dy, kz*z+dz)
static void run_up(hipStream_t st, Arena& ws, const float* wup, int O, int I, int kz,
                   Act in, Grid4 gin, unsigned short* dst, int dstCs, int dstOff,
                   Grid4 gout) {
  const int taps = 4 * kz;
  unsigned short* wt = ws.bf((size_t)taps * O * I);
  { long n = (long)taps * O * I;
    unsigned g = (unsigned)((n + THREADS - 1) / THREADS); if (g > 4096) g = 4096;
    prep_w_tap_kernel<<<g, THREADS, 0, st>>>(wup, wt, O, I, taps); }
  float* tmp = ws.f32(gout.sites() * (size_t)O);
  for (int dx = 0; dx < 2; ++dx)
    for (int dy = 0; dy < 2; ++dy)
      for (int dz = 0; dz < kz; ++dz) {
        int tap = (dx * 2 + dy) * kz + dz;
        launch_conv(st, mk(in.p + in.off, in.Cs, wt + (size_t)tap * O * I, I, I, O,
                           tmp, O, 0, gin, gin, gout,
                           1,1,1,1, 1,1,1,1, 0,0,0,0,
                           2,2,kz, dx,dy,dz));
      }
  cvt_slice(st, tmp, O, dst, dstCs, dstOff, gout.sites(), O);
}

extern "C" void kernel_launch(void* const* d_in, const int* in_sizes, int n_in,
                              void* d_out, int out_size, void* d_ws, size_t ws_size,
                              hipStream_t stream) {
  (void)in_sizes; (void)n_in; (void)out_size;
  Arena ws{(char*)d_ws, ws_size, 0};
  InCur cur{d_in, 0};

  const float* x = cur.next();
  StdbW b111 = read_stdb(cur, 16, 16, 16, false, 2);
  StdbW b112 = read_stdb(cur, 16, 16, 32, true, 2);
  StdbW b211 = read_stdb(cur, 32, 32, 32, false, 2);
  StdbW b212 = read_stdb(cur, 32, 32, 64, true, 2);
  StdbW b311 = read_stdb(cur, 64, 64, 64, false, 2);
  StdbW b312 = read_stdb(cur, 64, 64, 64, true, 2);
  StdbW b411 = read_stdb(cur, 64, 64, 64, false, 2);
  StdbW b412 = read_stdb(cur, 64, 64, 64, true, 1);
  StdbW b511 = read_stdb(cur, 64, 64, 64, false, 2);
  StdbW b512 = read_stdb(cur, 64, 64, 64, false, 2);
  const float* up5 = cur.next();
  StdbW b421 = read_stdb(cur, 128, 128, 64, false, 2);
  const float* up4 = cur.next();
  StdbW b321 = read_stdb(cur, 128, 128, 64, false, 2);
  const float* up3 = cur.next();
  StdbW b221 = read_stdb(cur, 128, 64, 64, false, 2);
  const float* up2 = cur.next();
  StdbW b121 = read_stdb(cur, 64, 32, 16, false, 2);

  Grid4 g0{64,64,16,4}, g1{32,32,8,4}, g2{16,16,4,4}, g3{8,8,2,4}, g4{4,4,2,4};

  // x: NCXYZT f32 -> channels-last bf16
  unsigned short* xbf = ws.bf(g0.sites() * 16);
  { dim3 grid((unsigned)((g0.sites() + THREADS - 1) / THREADS), 16);
    nchw_to_bf16_kernel<<<grid, dim3(THREADS), 0, stream>>>(x, g0.sites(), xbf, 16, 0); }

  // ---- encoder ----
  Act h = run_stdb(stream, ws, b111, Act{xbf, 16, 0, 16}, g0, nullptr, nullptr, nullptr);
  Act skip1{}, d2in{}; Grid4 pg;
  skip1 = run_stdb(stream, ws, b112, h, g0, &d2in, &pg, nullptr);
  Act d2 = run_stdb(stream, ws, b211, d2in, g1, nullptr, nullptr, nullptr);
  Act skip2{}, d3in{};
  skip2 = run_stdb(stream, ws, b212, d2, g1, &d3in, &pg, nullptr);
  Act d3 = run_stdb(stream, ws, b311, d3in, g2, nullptr, nullptr, nullptr);
  Act skip3{}, d4in{};
  skip3 = run_stdb(stream, ws, b312, d3, g2, &d4in, &pg, nullptr);
  Act d4 = run_stdb(stream, ws, b411, d4in, g3, nullptr, nullptr, nullptr);
  Act skip4{}, d5in{};
  skip4 = run_stdb(stream, ws, b412, d4, g3, &d5in, &pg, nullptr);
  Act d5 = run_stdb(stream, ws, b511, d5in, g4, nullptr, nullptr, nullptr);
  d5 = run_stdb(stream, ws, b512, d5, g4, nullptr, nullptr, nullptr);

  // ---- decoder ----
  unsigned short* u4in = ws.bf(g3.sites() * 128);
  run_up(stream, ws, up5, 64, 64, 1, d5, g4, u4in, 128, 0, g3);
  copy_slice(stream, skip4.p, skip4.Cs, skip4.off, u4in, 128, 64, g3.sites(), 64);
  Act u4 = run_stdb(stream, ws, b421, Act{u4in, 128, 0, 128}, g3, nullptr, nullptr, nullptr);

  unsigned short* u3in = ws.bf(g2.sites() * 128);
  run_up(stream, ws, up4, 64, 64, 2, u4, g3, u3in, 128, 0, g2);
  copy_slice(stream, skip3.p, skip3.Cs, skip3.off, u3in, 128, 64, g2.sites(), 64);
  Act u3 = run_stdb(stream, ws, b321, Act{u3in, 128, 0, 128}, g2, nullptr, nullptr, nullptr);

  unsigned short* u2in = ws.bf(g1.sites() * 128);
  run_up(stream, ws, up3, 64, 64, 2, u3, g2, u2in, 128, 0, g1);
  copy_slice(stream, skip2.p, skip2.Cs, skip2.off, u2in, 128, 64, g1.sites(), 64);
  Act u2 = run_stdb(stream, ws, b221, Act{u2in, 128, 0, 128}, g1, nullptr, nullptr, nullptr);

  unsigned short* u1in = ws.bf(g0.sites() * 64);
  run_up(stream, ws, up2, 32, 64, 2, u2, g1, u1in, 64, 0, g0);
  copy_slice(stream, skip1.p, skip1.Cs, skip1.off, u1in, 64, 32, g0.sites(), 32);
  (void)run_stdb(stream, ws, b121, Act{u1in, 64, 0, 64}, g0, nullptr, nullptr,
                 (float*)d_out);
}
